// SS2D_Enhanced_28406913696121
// MI455X (gfx1250) — matrix operand order repaired and checked by reading, not compiled, not run on hardware
//
#include <hip/hip_runtime.h>
#include <hip/hip_bf16.h>
#include <stdint.h>

typedef __attribute__((ext_vector_type(16))) __bf16 v16bf;
typedef __attribute__((ext_vector_type(8)))  __bf16 v8bf;
typedef __attribute__((ext_vector_type(8)))  float  v8f;

#define BM 128
#define BN 64
#define BK 32
#define APITCH 40   // bf16 elements per LDS row: 80B, 16B-aligned, conflict-free

__device__ __forceinline__ void async_ld16(uint32_t lds_off, const void* g) {
  // GLOBAL_LOAD_ASYNC_TO_LDS_B128: vdst = LDS byte addr (VGPR), vaddr = 64-bit global
  asm volatile("global_load_async_to_lds_b128 %0, %1, off"
               :: "v"(lds_off), "v"(g) : "memory");
}

__device__ __forceinline__ v16bf concat8(v8bf lo, v8bf hi) {
  return __builtin_shufflevector(lo, hi, 0, 1, 2, 3, 4, 5, 6, 7,
                                 8, 9, 10, 11, 12, 13, 14, 15);
}

__device__ __forceinline__ float silu_f(float v) {
  return v / (1.f + __expf(-v));
}

// ---------------------------------------------------------------------------
// Tiled transpose + bf16 quantize: x (B, C, L) f32 -> u (B, L, C) bf16
// ---------------------------------------------------------------------------
__global__ __launch_bounds__(256) void transpose_cl_lc_bf16(
    const float* __restrict__ x, __bf16* __restrict__ u, int C, int L)
{
  __shared__ float t[32][33];
  const int b  = blockIdx.z;
  const int l0 = blockIdx.x * 32, c0 = blockIdx.y * 32;
  const int tx = threadIdx.x & 31, ty = threadIdx.x >> 5;
  const float* xb = x + (size_t)b * C * L;
#pragma unroll
  for (int i = 0; i < 4; ++i) {
    int c = c0 + ty + i * 8;
    t[ty + i * 8][tx] = xb[(size_t)c * L + l0 + tx];
  }
  __syncthreads();
  __bf16* ub = u + (size_t)b * L * C;
#pragma unroll
  for (int i = 0; i < 4; ++i) {
    int l = l0 + ty + i * 8;
    ub[(size_t)l * C + c0 + tx] = (__bf16)t[tx][ty + i * 8];
  }
}

// ---------------------------------------------------------------------------
// Elementwise f32 -> bf16 (weights)
// ---------------------------------------------------------------------------
__global__ void f32_to_bf16_kernel(const float* __restrict__ s,
                                   __bf16* __restrict__ d, int n)
{
  int i = blockIdx.x * 256 + threadIdx.x;
  if (i < n) d[i] = (__bf16)s[i];
}

// ---------------------------------------------------------------------------
// C[M,N] f32 = A[M,K] bf16 (row major) * W[N,K]^T bf16, WMMA f32 accumulate.
// 256 threads (8 waves); block tile 128x64; each wave owns 2x2 16x16 subtiles
// (4 v_wmma per K-step). A/B tiles staged via async global->LDS (ASYNCcnt),
// double buffered; fragments are direct ds_load_b128 in the ISA 7.12.2 layout.
// ---------------------------------------------------------------------------
__global__ __launch_bounds__(256) void gemm_bf16_wmma(
    const __bf16* __restrict__ A, const __bf16* __restrict__ W,
    float* __restrict__ C, int M, int N, int K)
{
  __shared__ __align__(16) __bf16 sA[2][BM * APITCH];
  __shared__ __align__(16) __bf16 sB[2][BN * APITCH];
  __shared__ __align__(16) __bf16 sDummy[64];

  const int tid  = threadIdx.x;
  const int wave = tid >> 5;
  const int lane = tid & 31;
  const int m0 = blockIdx.x * BM;
  const int n0 = blockIdx.y * BN;

  const int mi = (wave & 3) * 2;      // M sub-tiles mi, mi+1   (8 total)
  const int nj = (wave >> 2) * 2;     // N sub-tiles nj, nj+1   (4 total)

  v8f acc00 = {}, acc01 = {}, acc10 = {}, acc11 = {};

  // zero tail rows of B buffers once (partial N tile); asyncs never touch them
  if (n0 + BN > N) {
    for (int i = tid; i < 2 * BN * APITCH; i += 256) ((__bf16*)sB)[i] = (__bf16)0.f;
  }
  __syncthreads();

  auto prefetch = [&](int kk, int buf) {
#pragma unroll
    for (int q = 0; q < 2; ++q) {            // A tile: 128 rows x 4 chunks
      int ch  = tid + q * 256;
      int row = ch >> 2, c8 = (ch & 3) * 8;
      async_ld16((uint32_t)(size_t)&sA[buf][row * APITCH + c8],
                 A + (size_t)(m0 + row) * K + kk + c8);
    }
    {                                        // B tile: 64 rows x 4 chunks
      int row = tid >> 2, c8 = (tid & 3) * 8;
      bool ok = (n0 + row) < N;              // wave-uniform issue via selects
      uint32_t dst = ok ? (uint32_t)(size_t)&sB[buf][row * APITCH + c8]
                        : (uint32_t)(size_t)&sDummy[0];
      const __bf16* g = ok ? (W + (size_t)(n0 + row) * K + kk + c8) : W;
      async_ld16(dst, g);
    }
  };

  prefetch(0, 0);
  const int iters = K / BK;
  const int half = lane >> 4, r = lane & 15;

  for (int it = 0; it < iters; ++it) {
    const int buf = it & 1;
    if (it + 1 < iters) {
      prefetch((it + 1) * BK, buf ^ 1);
      asm volatile("s_wait_asynccnt 3" ::: "memory");   // current tile's 3 done
    } else {
      asm volatile("s_wait_asynccnt 0" ::: "memory");
    }
    __syncthreads();

    // A fragments (16-bit A 16x32 layout): lane<16 -> {K0..7, K16..23},
    // lane>=16 -> {K8..15, K24..31}; each half is one 16B ds_load_b128.
    const __bf16* Ar0 = &sA[buf][(mi * 16 + r) * APITCH];
    const __bf16* Ar1 = &sA[buf][((mi + 1) * 16 + r) * APITCH];
    v16bf a0 = concat8(*(const v8bf*)(Ar0 + 8 * half),
                       *(const v8bf*)(Ar0 + 16 + 8 * half));
    v16bf a1 = concat8(*(const v8bf*)(Ar1 + 8 * half),
                       *(const v8bf*)(Ar1 + 16 + 8 * half));

    // B fragments (B 32x16 layout): lane<16 -> {K0..7, K16..23}, lane>=16 -> {K8..15, K24..31}.
    const __bf16* Br0 = &sB[buf][(nj * 16 + r) * APITCH];
    const __bf16* Br1 = &sB[buf][((nj + 1) * 16 + r) * APITCH];
    v16bf b0 = concat8(*(const v8bf*)(Br0 + 8 * half),
                       *(const v8bf*)(Br0 + 16 + 8 * half));
    v16bf b1 = concat8(*(const v8bf*)(Br1 + 8 * half),
                       *(const v8bf*)(Br1 + 16 + 8 * half));

    acc00 = __builtin_amdgcn_wmma_f32_16x16x32_bf16(false, a0, false, b0,
                                                    (short)0, acc00, false, false);
    acc01 = __builtin_amdgcn_wmma_f32_16x16x32_bf16(false, a0, false, b1,
                                                    (short)0, acc01, false, false);
    acc10 = __builtin_amdgcn_wmma_f32_16x16x32_bf16(false, a1, false, b0,
                                                    (short)0, acc10, false, false);
    acc11 = __builtin_amdgcn_wmma_f32_16x16x32_bf16(false, a1, false, b1,
                                                    (short)0, acc11, false, false);
    __syncthreads();
  }

  // C layout: lane<16 -> N=lane, M=j ; lane>=16 -> N=lane-16, M=8+j
  const int col0 = n0 + nj * 16 + r;
  const int col1 = col0 + 16;
  const int rb0 = m0 + mi * 16 + half * 8;
  const int rb1 = rb0 + 16;
#pragma unroll
  for (int j = 0; j < 8; ++j) {
    if (col0 < N) C[(size_t)(rb0 + j) * N + col0] = acc00[j];
    if (col1 < N) C[(size_t)(rb0 + j) * N + col1] = acc01[j];
    if (col0 < N) C[(size_t)(rb1 + j) * N + col0] = acc10[j];
    if (col1 < N) C[(size_t)(rb1 + j) * N + col1] = acc11[j];
  }
}

// ---------------------------------------------------------------------------
// Depthwise causal conv over L (4 taps) + SiLU, and dt = softplus(dt+bias)
// ---------------------------------------------------------------------------
__global__ void conv_dt_kernel(
    const float* __restrict__ zxbcdt, const float* __restrict__ conv_w,
    const float* __restrict__ conv_b, const float* __restrict__ dt_bias,
    float* __restrict__ xconv, float* __restrict__ dtb, int Mtot, int L)
{
  int idx = blockIdx.x * blockDim.x + threadIdx.x;
  if (idx >= Mtot * 516) return;
  int c = idx % 516;
  int m = idx / 516;           // m = b*L + l
  int l = m % L;
  if (c < 512) {
    float acc = conv_b[c];
#pragma unroll
    for (int k = 0; k < 4; ++k) {
      int dl = k - 3;
      if (l + dl >= 0)
        acc += zxbcdt[(size_t)(m + dl) * 1284 + 768 + c] * conv_w[c * 4 + k];
    }
    xconv[(size_t)m * 512 + c] = silu_f(acc);
  } else {
    int h = c - 512;
    float v = zxbcdt[(size_t)m * 1284 + 1280 + h] + dt_bias[h];
    dtb[(size_t)m * 4 + h] = (v > 20.f) ? v : log1pf(__expf(v));
  }
}

// ---------------------------------------------------------------------------
// Selective scan: one block per (b, head). State 64x128 in registers
// (32 f32/thread). Register-pipelined LDS staging of B/C/x per step.
// ---------------------------------------------------------------------------
__global__ __launch_bounds__(256) void scan_kernel(
    const float* __restrict__ xconv, const float* __restrict__ dtb,
    const float* __restrict__ A_log, const float* __restrict__ Dp,
    float* __restrict__ ys, int L)
{
  const int b = blockIdx.x >> 2;
  const int h = blockIdx.x & 3;
  const int tid = threadIdx.x;
  const int p = tid & 63;
  const int g = tid >> 6;          // 0..3, n-chunk
  const int n0 = g << 5;

  __shared__ __align__(16) float sB[128];
  __shared__ __align__(16) float sC[128];
  __shared__ float sX[64];
  __shared__ float sDt;
  __shared__ float part[256];

  const float Ah = -__expf(A_log[h]);
  const float Dh = Dp[h];

  float hreg[32];
#pragma unroll
  for (int i = 0; i < 32; ++i) hreg[i] = 0.f;

  const float* xcb = xconv + (size_t)b * L * 512;
  const float* dtbb = dtb + (size_t)b * L * 4 + h;
  float* ysb = ys + (size_t)b * L * 256 + h * 64;

  float rB = 0.f, rC = 0.f, rX = 0.f, rDt = 0.f;
  {
    const float* row = xcb;
    if (tid < 128) rB = row[256 + tid]; else rC = row[384 + (tid - 128)];
    if (tid < 64)  rX = row[h * 64 + tid];
    if (tid == 0)  rDt = dtbb[0];
  }

  for (int t = 0; t < L; ++t) {
    if (tid < 128) sB[tid] = rB; else sC[tid - 128] = rC;
    if (tid < 64)  sX[tid] = rX;
    if (tid == 0)  sDt = rDt;
    __syncthreads();

    if (t + 1 < L) {                     // prefetch next step into registers
      const float* row = xcb + (size_t)(t + 1) * 512;
      if (tid < 128) rB = row[256 + tid]; else rC = row[384 + (tid - 128)];
      if (tid < 64)  rX = row[h * 64 + tid];
      if (tid == 0)  rDt = dtbb[(size_t)(t + 1) * 4];
    }

    const float dtt  = sDt;
    const float dA   = __expf(dtt * Ah);
    const float coef = dtt * sX[p];
    float acc = 0.f;
    const float4* B4 = (const float4*)&sB[n0];
    const float4* C4 = (const float4*)&sC[n0];
#pragma unroll
    for (int i = 0; i < 8; ++i) {
      float4 bv = B4[i], cv = C4[i];
      hreg[4*i+0] = hreg[4*i+0] * dA + coef * bv.x; acc += hreg[4*i+0] * cv.x;
      hreg[4*i+1] = hreg[4*i+1] * dA + coef * bv.y; acc += hreg[4*i+1] * cv.y;
      hreg[4*i+2] = hreg[4*i+2] * dA + coef * bv.z; acc += hreg[4*i+2] * cv.z;
      hreg[4*i+3] = hreg[4*i+3] * dA + coef * bv.w; acc += hreg[4*i+3] * cv.w;
    }
    part[tid] = acc;
    __syncthreads();
    if (g == 0) {
      float y = part[p] + part[64 + p] + part[128 + p] + part[192 + p]
              + Dh * sX[p];
      ysb[(size_t)t * 256 + p] = y;
    }
    __syncthreads();   // protect sB/sC/sX/part before next step's stores
  }
}

// ---------------------------------------------------------------------------
// y = ys * silu(z); RMSNorm(256) * norm_w; yfinal = [silu(z0)*x0, y_norm] bf16
// One block per token (m = b*L + l).
// ---------------------------------------------------------------------------
__global__ __launch_bounds__(256) void gate_norm_kernel(
    const float* __restrict__ ys, const float* __restrict__ zxbcdt,
    const float* __restrict__ norm_w, __bf16* __restrict__ yfinal)
{
  const int m = blockIdx.x;
  const int c = threadIdx.x;
  __shared__ float red[256];
  const float* zrow = zxbcdt + (size_t)m * 1284;

  float y = ys[(size_t)m * 256 + c];
  float z = zrow[512 + c];
  y *= silu_f(z);
  red[c] = y * y;
  __syncthreads();
  for (int s = 128; s > 0; s >>= 1) {
    if (c < s) red[c] += red[c + s];
    __syncthreads();
  }
  float rs = rsqrtf(red[0] / 256.f + 1e-5f);
  float yn = y * rs * norm_w[c];

  float z0 = zrow[c], x0 = zrow[256 + c];
  __bf16* o = yfinal + (size_t)m * 512;
  o[c]       = (__bf16)(silu_f(z0) * x0);
  o[256 + c] = (__bf16)yn;
}

// ---------------------------------------------------------------------------
// SE block helpers
// ---------------------------------------------------------------------------
__global__ void zero_kernel(float* p, int n) {
  int i = blockIdx.x * 256 + threadIdx.x;
  if (i < n) p[i] = 0.f;
}

__global__ __launch_bounds__(256) void mean_partial_kernel(
    const float* __restrict__ outp, float* __restrict__ meanb, int L)
{
  const int b = blockIdx.x >> 4;
  const int chunk = blockIdx.x & 15;
  const int c = threadIdx.x;
  const int span = L / 16;
  const float* base = outp + ((size_t)b * L + chunk * span) * 256 + c;
  float s = 0.f;
  for (int i = 0; i < span; ++i) s += base[(size_t)i * 256];
  atomicAdd(&meanb[b * 256 + c], s);
}

__global__ __launch_bounds__(256) void se_kernel(
    const float* __restrict__ meanb,
    const float* __restrict__ w1, const float* __restrict__ b1,
    const float* __restrict__ w2, const float* __restrict__ b2,
    float* __restrict__ ses, int L)
{
  const int b = blockIdx.x, c = threadIdx.x;
  __shared__ float mrow[256], hbuf[64];
  mrow[c] = meanb[b * 256 + c] / (float)L;
  __syncthreads();
  if (c < 64) {
    float acc = b1[c];
    for (int k = 0; k < 256; ++k) acc += w1[c * 256 + k] * mrow[k];
    hbuf[c] = fmaxf(acc, 0.f);
  }
  __syncthreads();
  float acc = b2[c];
  for (int k = 0; k < 64; ++k) acc += w2[c * 64 + k] * hbuf[k];
  ses[b * 256 + c] = 1.f / (1.f + __expf(-acc));
}

// ---------------------------------------------------------------------------
// out (B,C,L) = outp (B,L,C) transposed * se scale, LDS-tiled
// ---------------------------------------------------------------------------
__global__ __launch_bounds__(256) void scale_transpose_kernel(
    const float* __restrict__ outp, const float* __restrict__ ses,
    float* __restrict__ out, int C, int L)
{
  __shared__ float t[32][33];
  const int b  = blockIdx.z;
  const int l0 = blockIdx.x * 32, c0 = blockIdx.y * 32;
  const int tx = threadIdx.x & 31, ty = threadIdx.x >> 5;
  const float* ob = outp + (size_t)b * L * C;
#pragma unroll
  for (int i = 0; i < 4; ++i) {
    int l = l0 + ty + i * 8;
    t[ty + i * 8][tx] = ob[(size_t)l * C + c0 + tx];
  }
  __syncthreads();
  float* db = out + (size_t)b * C * L;
#pragma unroll
  for (int i = 0; i < 4; ++i) {
    int c = c0 + ty + i * 8;
    db[(size_t)c * L + l0 + tx] = t[tx][ty + i * 8] * ses[b * 256 + c];
  }
}

// ---------------------------------------------------------------------------
extern "C" void kernel_launch(void* const* d_in, const int* in_sizes, int n_in,
                              void* d_out, int out_size, void* d_ws, size_t ws_size,
                              hipStream_t stream)
{
  (void)in_sizes; (void)n_in; (void)out_size; (void)ws_size;

  const float* x          = (const float*)d_in[0];
  const float* in_proj_w  = (const float*)d_in[1];
  const float* conv_w     = (const float*)d_in[2];
  const float* conv_b     = (const float*)d_in[3];
  const float* dt_bias    = (const float*)d_in[4];
  const float* A_log      = (const float*)d_in[5];
  const float* Dp         = (const float*)d_in[6];
  const float* norm_w     = (const float*)d_in[7];
  const float* out_proj_w = (const float*)d_in[8];
  const float* se_w1      = (const float*)d_in[9];
  const float* se_b1      = (const float*)d_in[10];
  const float* se_w2      = (const float*)d_in[11];
  const float* se_b2      = (const float*)d_in[12];

  const int B = 8, C = 256, L = 64 * 64;
  const int M = B * L;             // 32768 tokens
  const int Nin = 1284, Kin = 256;
  const int Nout = 256, Kout = 512;

  char* ws = (char*)d_ws;
  size_t off = 0;
  auto carve = [&](size_t bytes) -> void* {
    void* p = (void*)(ws + off);
    off += (bytes + 255) & ~(size_t)255;
    return p;
  };
  __bf16* u      = (__bf16*)carve((size_t)M * Kin * 2);    // (M,256) bf16
  __bf16* wbf1   = (__bf16*)carve((size_t)Nin * Kin * 2);  // in_proj bf16
  __bf16* wbf2   = (__bf16*)carve((size_t)Nout * Kout * 2);// out_proj bf16
  float*  zxbcdt = (float*) carve((size_t)M * Nin * 4);    // (M,1284)
  float*  xconv  = (float*) carve((size_t)M * 512 * 4);    // (M,512)
  float*  dtb    = (float*) carve((size_t)M * 4   * 4);    // (M,4)
  float*  ysb    = (float*) carve((size_t)M * 256 * 4);    // (M,256)
  __bf16* yfin   = (__bf16*)carve((size_t)M * 512 * 2);    // (M,512) bf16
  float*  outp   = (float*) carve((size_t)M * 256 * 4);    // (M,256)
  float*  meanb  = (float*) carve((size_t)B * 256 * 4);
  float*  sesb   = (float*) carve((size_t)B * 256 * 4);

  transpose_cl_lc_bf16<<<dim3(L / 32, C / 32, B), 256, 0, stream>>>(x, u, C, L);
  f32_to_bf16_kernel<<<(Nin * Kin + 255) / 256, 256, 0, stream>>>(
      in_proj_w, wbf1, Nin * Kin);
  f32_to_bf16_kernel<<<(Nout * Kout + 255) / 256, 256, 0, stream>>>(
      out_proj_w, wbf2, Nout * Kout);

  gemm_bf16_wmma<<<dim3(M / BM, (Nin + BN - 1) / BN), 256, 0, stream>>>(
      u, wbf1, zxbcdt, M, Nin, Kin);

  conv_dt_kernel<<<(M * 516 + 255) / 256, 256, 0, stream>>>(
      zxbcdt, conv_w, conv_b, dt_bias, xconv, dtb, M, L);

  scan_kernel<<<B * 4, 256, 0, stream>>>(xconv, dtb, A_log, Dp, ysb, L);

  gate_norm_kernel<<<M, 256, 0, stream>>>(ysb, zxbcdt, norm_w, yfin);

  gemm_bf16_wmma<<<dim3(M / BM, Nout / BN), 256, 0, stream>>>(
      yfin, wbf2, outp, M, Nout, Kout);

  zero_kernel<<<(B * 256 + 255) / 256, 256, 0, stream>>>(meanb, B * 256);
  mean_partial_kernel<<<B * 16, 256, 0, stream>>>(outp, meanb, L);
  se_kernel<<<B, 256, 0, stream>>>(meanb, se_w1, se_b1, se_w2, se_b2, sesb, L);

  scale_transpose_kernel<<<dim3(L / 32, C / 32, B), 256, 0, stream>>>(
      outp, sesb, (float*)d_out, C, L);
}
